// Decoder_39633958207518
// MI455X (gfx1250) — compile-verified, matrix operand order checked by reading, not run on hardware
//
#include <hip/hip_runtime.h>
#include <hip/hip_fp16.h>

typedef __attribute__((ext_vector_type(16))) _Float16 v16h;
typedef __attribute__((ext_vector_type(8)))  float    v8f;
typedef __attribute__((ext_vector_type(4)))  int      v4i;

#define BATCH  512
#define INPUT  128
#define HIDDEN 1024
#define SEQ    96

#define NT_X   (INPUT / 32)    // 4  K-tiles from x phase
#define NT_H   (HIDDEN / 32)   // 32 K-tiles from h phase
#define NT     (NT_X + NT_H)   // 36 total K-tiles

union Frag { uint4 q[2]; v16h h; };

// ---------------------------------------------------------------------------
// CDNA5 async global->LDS copy (ASYNCcnt-tracked, no VGPR round trip).
// Builtin signature (from hipcc diagnostics): src is AS(1) v4i*, dst AS(3).
// Guarded: falls back to a synchronous copy if the builtin is unavailable.
// ---------------------------------------------------------------------------
#if defined(__has_builtin)
#if __has_builtin(__builtin_amdgcn_global_load_async_to_lds_b128)
#define HAVE_ASYNC_LDS 1
#endif
#if __has_builtin(__builtin_amdgcn_s_wait_asynccnt)
#define HAVE_WAIT_ASYNC 1
#endif
#endif

__device__ inline void async_copy16(const _Float16* g, _Float16* l) {
#ifdef HAVE_ASYNC_LDS
  __builtin_amdgcn_global_load_async_to_lds_b128(
      (__attribute__((address_space(1))) v4i*)(_Float16*)g,
      (__attribute__((address_space(3))) v4i*)l, 0, 0);
#else
  *reinterpret_cast<uint4*>(l) = *reinterpret_cast<const uint4*>(g);
#endif
}

__device__ inline void wait_async_le2() {
#ifdef HAVE_WAIT_ASYNC
  __builtin_amdgcn_s_wait_asynccnt(2);
#else
  asm volatile("s_wait_asynccnt 0x2" ::: "memory");
#endif
}

__device__ inline void wait_async_le0() {
#ifdef HAVE_WAIT_ASYNC
  __builtin_amdgcn_s_wait_asynccnt(0);
#else
  asm volatile("s_wait_asynccnt 0x0" ::: "memory");
#endif
}

// f16 WMMA A/B fragment: lane holds 16 halves. Pair r (0..7) covers
// K = (r<4 ? 2r : 16+2(r-4)) + 8*lhi  -> two contiguous 16B groups:
// halves [8*lhi, 8*lhi+8) and [16+8*lhi, 16+8*lhi+8).
__device__ inline v16h load_frag_global(const _Float16* __restrict__ p,
                                        int lhi) {
  Frag f;
  p += 8 * lhi;
  f.q[0] = *reinterpret_cast<const uint4*>(p);
  f.q[1] = *reinterpret_cast<const uint4*>(p + 16);
  return f.h;
}

__device__ inline v16h load_frag_lds(const _Float16* As, int row, int lhi) {
  Frag f;
  const _Float16* p = As + row * 32 + 8 * lhi;
  f.q[0] = *reinterpret_cast<const uint4*>(p);
  f.q[1] = *reinterpret_cast<const uint4*>(p + 16);
  return f.h;
}

__device__ inline float sigmoidf_fast(float x) {
  return 1.0f / (1.0f + __expf(-x));
}

// ---------------------------------------------------------------------------
// Kernel 1: gates = x@Wih^T + h@Whh^T + bias; LSTM cell update.
// Block = 128 threads (4 waves). Wave w computes gate w for a 64x16 tile.
// K loop flattened over 36 tiles (4 from x, 32 from h); LDS A tile is
// double-buffered via async global->LDS loads; B fragments are pipelined
// one tile ahead in registers.  grid = (BATCH/64, HIDDEN/16) = (8, 64)
// ---------------------------------------------------------------------------
__global__ __launch_bounds__(128)
void lstm_gates_kernel(const _Float16* __restrict__ x,    // [512][128]
                       const _Float16* __restrict__ h,    // [512][1024]
                       const _Float16* __restrict__ Wih,  // [4096][128]
                       const _Float16* __restrict__ Whh,  // [4096][1024]
                       const float*    __restrict__ bias, // [4096] (b_ih+b_hh)
                       float*          __restrict__ c,    // [512][1024] in/out
                       _Float16*       __restrict__ hOut) // [512][1024]
{
  __shared__ __align__(16) _Float16 As[2][64 * 32];  // double-buffered A tile
  __shared__ float gbuf[4][64][16];                  // i,f,g,o pre-activations

  const int m0   = blockIdx.x * 64;
  const int n0   = blockIdx.y * 16;
  const int tid  = threadIdx.x;
  const int wv   = tid >> 5;              // wave id == gate id (i,f,g,o)
  const int lane = tid & 31;
  const int l15  = lane & 15;
  const int lhi  = lane >> 4;

  // Each thread stages two 16B chunks of the 64x32-half A tile.
  const int row0 = tid >> 2;              // rows 0..31
  const int row1 = row0 + 32;             // rows 32..63
  const int offh = (tid & 3) * 8;         // half offset within the row

  // Per-lane weight-row base pointers (row depends only on gate + column).
  const _Float16* bih = Wih + (size_t)(wv * HIDDEN + n0 + l15) * INPUT;
  const _Float16* bhh = Whh + (size_t)(wv * HIDDEN + n0 + l15) * HIDDEN;

  auto aSrc = [&](int t, int row) -> const _Float16* {
    return (t < NT_X)
        ? (x + (size_t)(m0 + row) * INPUT  + t * 32 + offh)
        : (h + (size_t)(m0 + row) * HIDDEN + (t - NT_X) * 32 + offh);
  };
  auto stage = [&](int t) {
    _Float16* dst = As[t & 1];
    async_copy16(aSrc(t, row0), dst + row0 * 32 + offh);
    async_copy16(aSrc(t, row1), dst + row1 * 32 + offh);
  };
  auto loadB = [&](int t) -> v16h {
    const _Float16* p = (t < NT_X) ? (bih + t * 32) : (bhh + (t - NT_X) * 32);
    return load_frag_global(p, lhi);
  };

  v8f acc[4] = {};                        // four 16x16 M-tiles

  stage(0);
  stage(1);
  v16h breg = loadB(0);

  for (int t = 0; t < NT; ++t) {
    if (t + 1 < NT) wait_async_le2();     // tile t resident, t+1 in flight
    else            wait_async_le0();     // last tile: nothing else in flight
    __syncthreads();

    v16h bnext = breg;
    if (t + 1 < NT) bnext = loadB(t + 1); // overlap B latency with WMMAs

    const _Float16* Ab = As[t & 1];
    #pragma unroll
    for (int mt = 0; mt < 4; ++mt) {
      v16h afrag = load_frag_lds(Ab, mt * 16 + l15, lhi);
      acc[mt] = __builtin_amdgcn_wmma_f32_16x16x32_f16(
          false, afrag, false, breg, (short)0, acc[mt], false, false);
    }
    __syncthreads();                      // all reads of buffer t&1 done
    if (t + 2 < NT) stage(t + 2);         // refill the buffer just freed
    breg = bnext;
  }

  // Bias (depends on column only) + publish gates to LDS for cross-wave mix.
  const float bv = bias[wv * HIDDEN + n0 + l15];
  #pragma unroll
  for (int t = 0; t < 4; ++t) {
    #pragma unroll
    for (int v = 0; v < 8; ++v) {
      gbuf[wv][t * 16 + v + 8 * lhi][l15] = acc[t][v] + bv;
    }
  }
  __syncthreads();

  // Elementwise LSTM cell: 64*16 = 1024 elements, 8 per thread.
  #pragma unroll
  for (int e = 0; e < 8; ++e) {
    int idx = tid * 8 + e;
    int ml = idx >> 4, nl = idx & 15;
    float iv = gbuf[0][ml][nl];
    float fv = gbuf[1][ml][nl];
    float gv = gbuf[2][ml][nl];
    float ov = gbuf[3][ml][nl];
    int gi = (m0 + ml) * HIDDEN + n0 + nl;
    float cold = c[gi];
    float cn = sigmoidf_fast(fv) * cold + sigmoidf_fast(iv) * tanhf(gv);
    float hn = sigmoidf_fast(ov) * tanhf(cn);
    c[gi] = cn;
    hOut[gi] = (_Float16)hn;
  }
}

// ---------------------------------------------------------------------------
// Kernel 2: y = h_new @ Wfc^T + b_fc ; emit y (f16) as next x, y[:,127] (f32)
// Block = 128 threads (4 waves); wave w owns M-tile w. grid = (8, 8).
// ---------------------------------------------------------------------------
__global__ __launch_bounds__(128)
void lstm_proj_kernel(const _Float16* __restrict__ h,    // [512][1024]
                      const _Float16* __restrict__ Wfc,  // [128][1024]
                      const float*    __restrict__ bfc,  // [128]
                      _Float16*       __restrict__ xOut, // [512][128]
                      float*          __restrict__ out,  // [512][96]
                      int step)
{
  const int wv   = threadIdx.x >> 5;
  const int lane = threadIdx.x & 31;
  const int l15  = lane & 15;
  const int lhi  = lane >> 4;
  const int m0   = blockIdx.x * 64 + wv * 16;
  const int n0   = blockIdx.y * 16;

  const _Float16* ap = h   + (size_t)(m0 + l15) * HIDDEN;
  const _Float16* bp = Wfc + (size_t)(n0 + l15) * HIDDEN;

  v8f acc = {};
  for (int kb = 0; kb < HIDDEN; kb += 32) {
    v16h af = load_frag_global(ap + kb, lhi);
    v16h bf = load_frag_global(bp + kb, lhi);
    acc = __builtin_amdgcn_wmma_f32_16x16x32_f16(
        false, af, false, bf, (short)0, acc, false, false);
  }

  const int j = n0 + l15;
  const float bv = bfc[j];
  #pragma unroll
  for (int v = 0; v < 8; ++v) {
    int m = m0 + v + 8 * lhi;
    float val = acc[v] + bv;
    xOut[m * INPUT + j] = (_Float16)val;
    if (j == INPUT - 1) out[m * SEQ + step] = val;
  }
}

// ---------------------------------------------------------------------------
// Prep kernels (run every launch; deterministic, inputs never mutated)
// ---------------------------------------------------------------------------
__global__ void cvt_f32_to_f16(const float* __restrict__ in,
                               _Float16* __restrict__ o, int n) {
  int i = blockIdx.x * blockDim.x + threadIdx.x;
  if (i < n) o[i] = (_Float16)in[i];
}

__global__ void bias_combine(const float* __restrict__ a,
                             const float* __restrict__ b,
                             float* __restrict__ o, int n) {
  int i = blockIdx.x * blockDim.x + threadIdx.x;
  if (i < n) o[i] = a[i] + b[i];
}

__global__ void copy_f32(const float* __restrict__ in,
                         float* __restrict__ o, int n) {
  int i = blockIdx.x * blockDim.x + threadIdx.x;
  if (i < n) o[i] = in[i];
}

// ---------------------------------------------------------------------------
extern "C" void kernel_launch(void* const* d_in, const int* in_sizes, int n_in,
                              void* d_out, int out_size, void* d_ws, size_t ws_size,
                              hipStream_t stream) {
  const float* xt     = (const float*)d_in[0]; // [512][1][128]
  const float* hidden = (const float*)d_in[1]; // [1][512][1024]
  const float* cell   = (const float*)d_in[2]; // [1][512][1024]
  const float* W_ih   = (const float*)d_in[3]; // [4096][128]
  const float* W_hh   = (const float*)d_in[4]; // [4096][1024]
  const float* b_ih   = (const float*)d_in[5]; // [4096]
  const float* b_hh   = (const float*)d_in[6]; // [4096]
  const float* W_fc   = (const float*)d_in[7]; // [128][1024]
  const float* b_fc   = (const float*)d_in[8]; // [128]
  float* out = (float*)d_out;                  // [512][96][1]

  // Workspace layout (256B-aligned sections), ~13 MB total.
  char* ws = (char*)d_ws;
  size_t off = 0;
  auto alloc = [&](size_t bytes) {
    char* p = ws + off;
    off = (off + bytes + 255) & ~(size_t)255;
    return p;
  };
  _Float16* Wih_h = (_Float16*)alloc((size_t)4 * HIDDEN * INPUT  * 2);
  _Float16* Whh_h = (_Float16*)alloc((size_t)4 * HIDDEN * HIDDEN * 2);
  _Float16* Wfc_h = (_Float16*)alloc((size_t)INPUT * HIDDEN * 2);
  float*    bias  = (float*)   alloc((size_t)4 * HIDDEN * 4);
  _Float16* x_h   = (_Float16*)alloc((size_t)BATCH * INPUT  * 2);
  _Float16* h_h   = (_Float16*)alloc((size_t)BATCH * HIDDEN * 2);
  float*    c_ws  = (float*)   alloc((size_t)BATCH * HIDDEN * 4);
  (void)ws_size; (void)in_sizes; (void)n_in; (void)out_size;

  auto launch_cvt = [&](const float* src, _Float16* dst, int n) {
    cvt_f32_to_f16<<<(n + 255) / 256, 256, 0, stream>>>(src, dst, n);
  };
  launch_cvt(W_ih,   Wih_h, 4 * HIDDEN * INPUT);
  launch_cvt(W_hh,   Whh_h, 4 * HIDDEN * HIDDEN);
  launch_cvt(W_fc,   Wfc_h, INPUT * HIDDEN);
  launch_cvt(xt,     x_h,   BATCH * INPUT);
  launch_cvt(hidden, h_h,   BATCH * HIDDEN);
  bias_combine<<<(4 * HIDDEN + 255) / 256, 256, 0, stream>>>(b_ih, b_hh, bias,
                                                             4 * HIDDEN);
  copy_f32<<<(BATCH * HIDDEN + 255) / 256, 256, 0, stream>>>(cell, c_ws,
                                                             BATCH * HIDDEN);

  dim3 gGrid(BATCH / 64, HIDDEN / 16);   // (8, 64)
  dim3 pGrid(BATCH / 64, INPUT / 16);    // (8, 8)
  for (int t = 0; t < SEQ; ++t) {
    lstm_gates_kernel<<<gGrid, 128, 0, stream>>>(x_h, h_h, Wih_h, Whh_h, bias,
                                                 c_ws, h_h);
    lstm_proj_kernel<<<pGrid, 128, 0, stream>>>(h_h, Wfc_h, b_fc, x_h, out, t);
  }
}
